// AuxiliaryModelWithRBERT_16234976379133
// MI455X (gfx1250) — matrix operand order repaired and checked by reading, not compiled, not run on hardware
//
#include <hip/hip_runtime.h>

#define B_ 1024
#define S_ 256
#define H_ 1024
#define NL 30

typedef __attribute__((ext_vector_type(16))) _Float16 v16h;
typedef __attribute__((ext_vector_type(8)))  float    v8f;

// ---------------------------------------------------------------------------
// Kernel 1: fused 4-mask entity average over sequence_output (1 GiB, 1 pass).
// Outputs: e3,e4 as f32 [B,H]; tanh(e1),tanh(e2) as f16 [B,H] (GEMM A-operand).
// ---------------------------------------------------------------------------
__global__ void entity_avg_kernel(const float* __restrict__ seq,
                                  const int* __restrict__ m1, const int* __restrict__ m2,
                                  const int* __restrict__ m3, const int* __restrict__ m4,
                                  _Float16* __restrict__ t1, _Float16* __restrict__ t2,
                                  float* __restrict__ e3, float* __restrict__ e4) {
  __shared__ float sm[4][S_];
  const int b = blockIdx.x;
  const int t = threadIdx.x;               // 256 threads
  sm[0][t] = (float)m1[b * S_ + t];
  sm[1][t] = (float)m2[b * S_ + t];
  sm[2][t] = (float)m3[b * S_ + t];
  sm[3][t] = (float)m4[b * S_ + t];
  __syncthreads();

  float acc[4][4] = {};
  float len[4] = {};
  const size_t rowbase = (size_t)b * S_ * H_ + 4 * t;
  for (int s = 0; s < S_; ++s) {
    const float* p = seq + rowbase + (size_t)s * H_;
    if (s + 8 < S_) __builtin_prefetch(seq + rowbase + (size_t)(s + 8) * H_, 0, 1);
    const float4 v = *reinterpret_cast<const float4*>(p);
    const float vv[4] = {v.x, v.y, v.z, v.w};
    #pragma unroll
    for (int e = 0; e < 4; ++e) {
      const float m = sm[e][s];
      len[e] += m;
      #pragma unroll
      for (int i = 0; i < 4; ++i) acc[e][i] += m * vv[i];
    }
  }
  const size_t o = (size_t)b * H_ + 4 * t;
  const float i1 = 1.0f / len[0], i2 = 1.0f / len[1];
  const float i3 = 1.0f / len[2], i4 = 1.0f / len[3];
  union { _Float16 h[4]; uint2 u; } p1, p2;
  float4 o3, o4;
  #pragma unroll
  for (int i = 0; i < 4; ++i) {
    p1.h[i] = (_Float16)tanhf(acc[0][i] * i1);
    p2.h[i] = (_Float16)tanhf(acc[1][i] * i2);
  }
  o3.x = acc[2][0] * i3; o3.y = acc[2][1] * i3; o3.z = acc[2][2] * i3; o3.w = acc[2][3] * i3;
  o4.x = acc[3][0] * i4; o4.y = acc[3][1] * i4; o4.z = acc[3][2] * i4; o4.w = acc[3][3] * i4;
  *reinterpret_cast<uint2*>(t1 + o) = p1.u;
  *reinterpret_cast<uint2*>(t2 + o) = p2.u;
  *reinterpret_cast<float4*>(e3 + o) = o3;
  *reinterpret_cast<float4*>(e4 + o) = o4;
}

// ---------------------------------------------------------------------------
// Kernel 2: tanh + f32->f16 for pooled_output
// ---------------------------------------------------------------------------
__global__ void tanh_f16_kernel(const float* __restrict__ in, _Float16* __restrict__ out) {
  const int i = blockIdx.x * blockDim.x + threadIdx.x;
  out[i] = (_Float16)tanhf(in[i]);
}

// ---------------------------------------------------------------------------
// Kernel 3: W [K,N] f32 row-major  ->  Wt [N,K] f16 (so B-fragments load
// contiguously: lane n holds column n, K ascending).
// ---------------------------------------------------------------------------
__global__ void transpose_w_kernel(const float* __restrict__ W, _Float16* __restrict__ Wt,
                                   int K, int N) {
  __shared__ float tile[32][33];
  const int nb = blockIdx.x * 32, kb = blockIdx.y * 32;
  const int tx = threadIdx.x, ty = threadIdx.y;           // 32 x 8
  #pragma unroll
  for (int i = 0; i < 32; i += 8)
    tile[ty + i][tx] = W[(size_t)(kb + ty + i) * N + nb + tx];
  __syncthreads();
  #pragma unroll
  for (int i = 0; i < 32; i += 8)
    Wt[(size_t)(nb + ty + i) * K + kb + tx] = (_Float16)tile[tx][ty + i];
}

// ---------------------------------------------------------------------------
// Kernel 4: WMMA GEMM  C[M,N] = A[M,K](f16) * Wt[N,K]^T(f16) + bias[N]  (f32 acc)
// One wave per 16x16 C tile, 8 waves/block. Fragment layouts used here (contiguous k, A and B matched):
//   A slot s (lane half h): k = 16h + s  -> two 16B loads
//   B lane = column (lane%16), lane half selects K 0-15 / 16-31 -> 32B load
//   C vgpr r: rows r (lanes 0-15) and r+8 (lanes 16-31), col = lane%16
// ---------------------------------------------------------------------------
__global__ void wmma_gemm_kernel(const _Float16* __restrict__ A,
                                 const _Float16* __restrict__ Bt,
                                 const float* __restrict__ bias,
                                 float* __restrict__ C, int M, int N, int K) {
  const int lane = threadIdx.x & 31;
  const int wid  = threadIdx.x >> 5;
  const int m0 = blockIdx.x * 16;
  const int n0 = blockIdx.y * 128 + wid * 16;
  const int half = lane >> 4;
  const int lr   = lane & 15;

  const _Float16* arow = A  + (size_t)(m0 + lr) * K;
  const _Float16* brow = Bt + (size_t)(n0 + lr) * K + 16 * half;

  v8f c = {};
  for (int k0 = 0; k0 < K; k0 += 32) {
    union { uint4 q[2]; v16h h; } ua, ub;
    ua.q[0] = *reinterpret_cast<const uint4*>(arow + k0 + 16 * half);      // K 0-7 of half
    ua.q[1] = *reinterpret_cast<const uint4*>(arow + k0 + 16 * half + 8);  // K 8-15 of half
    ub.q[0] = *reinterpret_cast<const uint4*>(brow + k0);                  // K 0-7 of half
    ub.q[1] = *reinterpret_cast<const uint4*>(brow + k0 + 8);              // K 8-15 of half
    c = __builtin_amdgcn_wmma_f32_16x16x32_f16(false, ua.h, false, ub.h,
                                               (short)0, c, false, false);
  }
  const float bv = bias[n0 + lr];
  #pragma unroll
  for (int r = 0; r < 8; ++r)
    C[(size_t)(m0 + r + 8 * half) * N + n0 + lr] = c[r] + bv;
}

// ---------------------------------------------------------------------------
// Kernel 5: head. Build xt = tanh(concat) in LDS, 62 dot products over K=5120,
// route via argmax(binary), write binary logits + selected label logits.
// ---------------------------------------------------------------------------
__global__ void head_kernel(const float* __restrict__ pf, const float* __restrict__ e1f,
                            const float* __restrict__ e2f, const float* __restrict__ e3,
                            const float* __restrict__ e4,
                            const float* __restrict__ Wb, const float* __restrict__ bb,
                            const float* __restrict__ W0, const float* __restrict__ b0,
                            const float* __restrict__ W1, const float* __restrict__ b1,
                            float* __restrict__ out_bin, float* __restrict__ out_log) {
  __shared__ float xt[5 * H_];
  __shared__ float res[62];
  const int b = blockIdx.x;
  const size_t ro = (size_t)b * H_;
  for (int j = threadIdx.x; j < H_; j += blockDim.x) {
    xt[j]          = tanhf(pf [ro + j]);
    xt[H_ + j]     = tanhf(e1f[ro + j]);
    xt[2 * H_ + j] = tanhf(e2f[ro + j]);
    xt[3 * H_ + j] = tanhf(e3 [ro + j]);
    xt[4 * H_ + j] = tanhf(e4 [ro + j]);
  }
  __syncthreads();
  const int j = threadIdx.x;
  if (j < 62) {
    const float* Wp; int nc, col; float acc;
    if (j < 2)       { Wp = Wb; nc = 2;  col = j;      acc = bb[col]; }
    else if (j < 32) { Wp = W0; nc = NL; col = j - 2;  acc = b0[col]; }
    else             { Wp = W1; nc = NL; col = j - 32; acc = b1[col]; }
    for (int k = 0; k < 5 * H_; ++k) acc += xt[k] * Wp[(size_t)k * nc + col];
    res[j] = acc;
  }
  __syncthreads();
  if (j < 2) out_bin[b * 2 + j] = res[j];
  const int route = (res[1] > res[0]) ? 1 : 0;   // argmax, ties -> 0
  if (j < NL) out_log[b * NL + j] = res[(route ? 32 : 2) + j];
}

// ---------------------------------------------------------------------------
extern "C" void kernel_launch(void* const* d_in, const int* in_sizes, int n_in,
                              void* d_out, int out_size, void* d_ws, size_t ws_size,
                              hipStream_t stream) {
  const float* seq    = (const float*)d_in[0];
  const float* pooled = (const float*)d_in[1];
  const int*   m1     = (const int*)d_in[2];
  const int*   m2     = (const int*)d_in[3];
  const int*   m3     = (const int*)d_in[4];
  const int*   m4     = (const int*)d_in[5];
  const float* Wcls   = (const float*)d_in[6];
  const float* bcls   = (const float*)d_in[7];
  const float* Went   = (const float*)d_in[8];
  const float* bent   = (const float*)d_in[9];
  const float* Wbin   = (const float*)d_in[10];
  const float* bbin   = (const float*)d_in[11];
  const float* W0     = (const float*)d_in[12];
  const float* b0     = (const float*)d_in[13];
  const float* W1     = (const float*)d_in[14];
  const float* b1     = (const float*)d_in[15];

  // workspace layout
  char* w = (char*)d_ws;
  const size_t BH2 = (size_t)B_ * H_ * sizeof(_Float16);   // 2 MiB
  const size_t BH4 = (size_t)B_ * H_ * sizeof(float);      // 4 MiB
  _Float16* t_pooled = (_Float16*)(w);             w += BH2;
  _Float16* t_e1     = (_Float16*)(w);             w += BH2;
  _Float16* t_e2     = (_Float16*)(w);             w += BH2;
  float*    e3       = (float*)(w);                w += BH4;
  float*    e4       = (float*)(w);                w += BH4;
  _Float16* Wcls_t   = (_Float16*)(w);             w += BH2;  // H*H f16
  _Float16* Went_t   = (_Float16*)(w);             w += BH2;
  float*    pooled_fc = (float*)(w);               w += BH4;
  float*    e1_fc     = (float*)(w);               w += BH4;
  float*    e2_fc     = (float*)(w);               w += BH4;

  float* out_bin = (float*)d_out;            // [B,2]
  float* out_log = (float*)d_out + B_ * 2;   // [B,30]

  // Stage 1: fused entity averages (single 1 GiB pass) + pooled tanh->f16 + W transposes
  entity_avg_kernel<<<B_, S_, 0, stream>>>(seq, m1, m2, m3, m4, t_e1, t_e2, e3, e4);
  tanh_f16_kernel<<<(B_ * H_) / 256, 256, 0, stream>>>(pooled, t_pooled);
  dim3 tb(32, 8), tg(H_ / 32, H_ / 32);
  transpose_w_kernel<<<tg, tb, 0, stream>>>(Wcls, Wcls_t, H_, H_);
  transpose_w_kernel<<<tg, tb, 0, stream>>>(Went, Went_t, H_, H_);

  // Stage 2: three [1024x1024x1024] WMMA GEMMs (f16 in, f32 acc, fused bias)
  dim3 gg(H_ / 16, H_ / 128);   // M/16 x N/128, 8 waves (16x16 tiles) per block
  wmma_gemm_kernel<<<gg, 256, 0, stream>>>(t_pooled, Wcls_t, bcls, pooled_fc, B_, H_, H_);
  wmma_gemm_kernel<<<gg, 256, 0, stream>>>(t_e1,     Went_t, bent, e1_fc,     B_, H_, H_);
  wmma_gemm_kernel<<<gg, 256, 0, stream>>>(t_e2,     Went_t, bent, e2_fc,     B_, H_, H_);

  // Stage 3: head (binary logits, routed label logits)
  head_kernel<<<B_, 128, 0, stream>>>(pooled_fc, e1_fc, e2_fc, e3, e4,
                                      Wbin, bbin, W0, b0, W1, b1, out_bin, out_log);
}
